// Grapher_43319040147862
// MI455X (gfx1250) — compile-verified, matrix operand order checked by reading, not run on hardware
//
#include <hip/hip_runtime.h>
#include <hip/hip_bf16.h>
#include <math.h>

typedef __attribute__((ext_vector_type(16))) __bf16 v16bf;
typedef __attribute__((ext_vector_type(8)))  __bf16 v8bf;
typedef __attribute__((ext_vector_type(8)))  float  v8f;

#define LDN 40   // LDS row stride in bf16 elements (80B: 16B-aligned, conflict-friendly)

__device__ __forceinline__ v8f wmma_bf16(v16bf a, v16bf b, v8f c) {
    return __builtin_amdgcn_wmma_f32_16x16x32_bf16(false, a, false, b, (short)0, c, false, false);
}

// Build a 16x32 (or 32x16-as-B) bf16 fragment from LDS, per CDNA5 WMMA layout:
// lane<16: row = lane,   K = {0..7} in elems 0..7, {16..23} in elems 8..15
// lane>=16: row = lane-16, K base +8 -> {8..15} and {24..31}
__device__ __forceinline__ v16bf frag_ld(const __bf16* base, int lane) {
    const __bf16* p = base + (lane & 15) * LDN + ((lane >> 4) << 3);
    v8bf lo8 = *(const v8bf*)(p);
    v8bf hi8 = *(const v8bf*)(p + 16);
    return __builtin_shufflevector(lo8, hi8, 0,1,2,3,4,5,6,7,8,9,10,11,12,13,14,15);
}

__device__ __forceinline__ void cvt_store(float f, __bf16* hi, __bf16* lo, int idx) {
    __bf16 h = (__bf16)f;
    hi[idx] = h;
    lo[idx] = (__bf16)(f - (float)h);
}

template<int KDIM>
__device__ __forceinline__ void fill_A(const float* __restrict__ A, int M, int rowBase, int k0,
                                       __bf16* hi, __bf16* lo, int tid) {
    #pragma unroll
    for (int it = 0; it < 4; ++it) {       // 128 rows x 32 k = 1024 float4 chunks / 256 thr
        int c = tid + it * 256;
        int r = c >> 3, kc = (c & 7) << 2;
        int gr = rowBase + r;
        float4 v = {0.f, 0.f, 0.f, 0.f};
        if (gr < M) v = *(const float4*)(A + (size_t)gr * KDIM + k0 + kc);
        int o = r * LDN + kc;
        cvt_store(v.x, hi, lo, o);   cvt_store(v.y, hi, lo, o + 1);
        cvt_store(v.z, hi, lo, o + 2); cvt_store(v.w, hi, lo, o + 3);
    }
}

template<int KDIM>
__device__ __forceinline__ void fill_B(const float* __restrict__ W, int colBase, int k0,
                                       __bf16* hi, __bf16* lo, int tid) {
    #pragma unroll
    for (int it = 0; it < 2; ++it) {       // 64 rows x 32 k = 512 float4 chunks / 256 thr
        int c = tid + it * 256;
        int r = c >> 3, kc = (c & 7) << 2;
        float4 v = *(const float4*)(W + (size_t)(colBase + r) * KDIM + k0 + kc);
        int o = r * LDN + kc;
        cvt_store(v.x, hi, lo, o);   cvt_store(v.y, hi, lo, o + 1);
        cvt_store(v.z, hi, lo, o + 2); cvt_store(v.w, hi, lo, o + 3);
    }
}

// C = A1 @ W1^T [+ A2 @ W2^T] + bias, optional erf-GELU, optional column sum/sumsq stats.
// A: [M x KDIM] fp32 row-major, W: [NCOLS x KDIM] fp32 row-major.
// Split-bf16 (3x WMMA) for ~fp32 accuracy; fp32 accumulate.
template<int KDIM, int NCOLS, bool HAS2, bool GELU_OUT, bool STATS>
__global__ __launch_bounds__(256)
void gemm_kernel(const float* __restrict__ A1, const float* __restrict__ W1,
                 const float* __restrict__ A2, const float* __restrict__ W2,
                 const float* __restrict__ bias,
                 float* __restrict__ Cout, float* __restrict__ stats, int M)
{
    __shared__ __align__(16) __bf16 sAhi[2][128 * LDN], sAlo[2][128 * LDN];
    __shared__ __align__(16) __bf16 sBhi[2][64 * LDN],  sBlo[2][64 * LDN];

    const int tid = threadIdx.x, lane = tid & 31, w = tid >> 5;
    const int rowBase = blockIdx.x * 128;
    const int colBase = blockIdx.y * 64;
    const int mOff = (w & 3) * 32;          // 4 waves along M (128)
    const int nOff = (w >> 2) * 32;         // 2 waves along N (64)
    constexpr int NSRC = HAS2 ? 2 : 1;

    v8f acc[2][2] = {};

    for (int k0 = 0; k0 < KDIM; k0 += 32) {
        __syncthreads();
        fill_A<KDIM>(A1, M, rowBase, k0, sAhi[0], sAlo[0], tid);
        fill_B<KDIM>(W1, colBase, k0, sBhi[0], sBlo[0], tid);
        if constexpr (HAS2) {
            fill_A<KDIM>(A2, M, rowBase, k0, sAhi[1], sAlo[1], tid);
            fill_B<KDIM>(W2, colBase, k0, sBhi[1], sBlo[1], tid);
        }
        __syncthreads();

        #pragma unroll
        for (int s = 0; s < NSRC; ++s) {
            v16bf ah[2], al[2], bh[2], bl[2];
            #pragma unroll
            for (int i = 0; i < 2; ++i) {
                ah[i] = frag_ld(&sAhi[s][(mOff + 16 * i) * LDN], lane);
                al[i] = frag_ld(&sAlo[s][(mOff + 16 * i) * LDN], lane);
                bh[i] = frag_ld(&sBhi[s][(nOff + 16 * i) * LDN], lane);
                bl[i] = frag_ld(&sBlo[s][(nOff + 16 * i) * LDN], lane);
            }
            #pragma unroll
            for (int i = 0; i < 2; ++i)
                #pragma unroll
                for (int j = 0; j < 2; ++j) {
                    acc[i][j] = wmma_bf16(ah[i], bh[j], acc[i][j]);  // hi*hi
                    acc[i][j] = wmma_bf16(ah[i], bl[j], acc[i][j]);  // hi*lo
                    acc[i][j] = wmma_bf16(al[i], bh[j], acc[i][j]);  // lo*hi
                }
        }
    }

    // Epilogue. D layout: VGPR v, lane l -> row = v + 8*(l>>4), col = l&15 within tile.
    #pragma unroll
    for (int i = 0; i < 2; ++i) {
        int gr0 = rowBase + mOff + 16 * i;
        if (gr0 >= M) continue;            // M % 16 == 0 -> tiles are all-or-nothing valid
        #pragma unroll
        for (int j = 0; j < 2; ++j) {
            int col = colBase + nOff + 16 * j + (lane & 15);
            float b = bias[col];
            float s = 0.f, ss = 0.f;
            #pragma unroll
            for (int v = 0; v < 8; ++v) {
                int row = gr0 + v + 8 * (lane >> 4);
                float val = acc[i][j][v] + b;
                if constexpr (GELU_OUT)
                    val = 0.5f * val * (1.0f + erff(val * 0.7071067811865476f));
                Cout[(size_t)row * NCOLS + col] = val;
                s += val; ss += val * val;
            }
            if constexpr (STATS) {
                s  += __shfl_down(s, 16);   // lanes j and j+16 hold the same column
                ss += __shfl_down(ss, 16);
                if (lane < 16) {
                    unsafeAtomicAdd(&stats[col], s);
                    unsafeAtomicAdd(&stats[NCOLS + col], ss);
                }
            }
        }
    }
}

__global__ void zero_kernel(float* __restrict__ p, size_t n) {
    size_t i = (size_t)blockIdx.x * blockDim.x + threadIdx.x;
    size_t st = (size_t)gridDim.x * blockDim.x;
    for (; i < n; i += st) p[i] = 0.f;
}

__global__ void finalize_bn(const float* __restrict__ stats, const float* __restrict__ gamma,
                            const float* __restrict__ beta, float* __restrict__ scale,
                            float* __restrict__ shift, int C, float invN) {
    int c = blockIdx.x * blockDim.x + threadIdx.x;
    if (c < C) {
        float mean = stats[c] * invN;
        float var  = stats[C + c] * invN - mean * mean;
        float sc   = gamma[c] * rsqrtf(var + 1e-5f);
        scale[c] = sc;
        shift[c] = beta[c] - mean * sc;
    }
}

__global__ void affine_inplace(float* __restrict__ x, const float* __restrict__ scale,
                               const float* __restrict__ shift, size_t n4, int Cdiv4) {
    size_t i = (size_t)blockIdx.x * blockDim.x + threadIdx.x;
    size_t st = (size_t)gridDim.x * blockDim.x;
    float4* p = (float4*)x;
    for (; i < n4; i += st) {
        int c = (int)(i % Cdiv4) * 4;
        float4 v = p[i];
        v.x = v.x * scale[c + 0] + shift[c + 0];
        v.y = v.y * scale[c + 1] + shift[c + 1];
        v.z = v.z * scale[c + 2] + shift[c + 2];
        v.w = v.w * scale[c + 3] + shift[c + 3];
        p[i] = v;
    }
}

// agg[dst[e], :] += h[src[e], :]   (H = 256). 64 threads/edge, float4 each.
__global__ __launch_bounds__(256)
void scatter_kernel(const float* __restrict__ h, const int* __restrict__ src,
                    const int* __restrict__ dst, float* __restrict__ agg, int E) {
    int sub  = threadIdx.x & 63;   // column-group 0..63 (x4 floats)
    int eoff = threadIdx.x >> 6;   // 4 edges per block iteration
    for (size_t e = (size_t)blockIdx.x * 4 + eoff; e < (size_t)E; e += (size_t)gridDim.x * 4) {
        int s = src[e], d = dst[e];
        float4 v = *(const float4*)(h + (size_t)s * 256 + sub * 4);
        float* p = agg + (size_t)d * 256 + sub * 4;
        unsafeAtomicAdd(p + 0, v.x);
        unsafeAtomicAdd(p + 1, v.y);
        unsafeAtomicAdd(p + 2, v.z);
        unsafeAtomicAdd(p + 3, v.w);
    }
}

__global__ void final_residual(const float* __restrict__ ypre, const float* __restrict__ x,
                               const float* __restrict__ scale, const float* __restrict__ shift,
                               float* __restrict__ out, size_t n4, int Cdiv4) {
    size_t i = (size_t)blockIdx.x * blockDim.x + threadIdx.x;
    size_t st = (size_t)gridDim.x * blockDim.x;
    const float4* py = (const float4*)ypre;
    const float4* px = (const float4*)x;
    float4* po = (float4*)out;
    for (; i < n4; i += st) {
        int c = (int)(i % Cdiv4) * 4;
        float4 v = py[i], xv = px[i];
        v.x = v.x * scale[c + 0] + shift[c + 0] + xv.x;
        v.y = v.y * scale[c + 1] + shift[c + 1] + xv.y;
        v.z = v.z * scale[c + 2] + shift[c + 2] + xv.z;
        v.w = v.w * scale[c + 3] + shift[c + 3] + xv.w;
        po[i] = v;
    }
}

extern "C" void kernel_launch(void* const* d_in, const int* in_sizes, int n_in,
                              void* d_out, int out_size, void* d_ws, size_t ws_size,
                              hipStream_t stream) {
    const float* x      = (const float*)d_in[0];
    const int*   ei     = (const int*)  d_in[1];
    const float* w1     = (const float*)d_in[2];
    const float* b1     = (const float*)d_in[3];
    const float* gamma1 = (const float*)d_in[4];
    const float* beta1  = (const float*)d_in[5];
    const float* w_root = (const float*)d_in[6];
    const float* w_nei  = (const float*)d_in[7];
    const float* b_gc   = (const float*)d_in[8];
    const float* w2     = (const float*)d_in[9];
    const float* b2     = (const float*)d_in[10];
    const float* gamma2 = (const float*)d_in[11];
    const float* beta2  = (const float*)d_in[12];

    const int D = 128, H = 256;
    const int N = in_sizes[0] / D;
    const int E = in_sizes[1] / 2;
    const size_t NH = (size_t)N * H, ND = (size_t)N * D;

    float* ws     = (float*)d_ws;
    float* h      = ws;                 // [N,256] (h_pre -> h in place); later aliased as y_pre
    float* agg    = ws + NH;            // [N,256]
    float* g      = ws + 2 * NH;        // [N,256]
    float* stats  = ws + 3 * NH;        // small stats arena
    float* sum1   = stats;              // [256] + sumsq [256] contiguous
    float* scale1 = stats + 512;
    float* shift1 = stats + 768;
    float* sum2   = stats + 1024;       // [128] + sumsq [128] contiguous
    float* scale2 = stats + 1280;
    float* shift2 = stats + 1408;
    float* ypre   = ws;                 // alias h (dead after GraphConv GEMM)

    const int gridM = (N + 127) / 128;
    const int* src = ei;
    const int* dst = ei + E;

    // deterministic re-init (harness poisons ws once, never between replays)
    zero_kernel<<<2048, 256, 0, stream>>>(agg, NH);
    zero_kernel<<<8, 256, 0, stream>>>(stats, 2048);

    // fc1 (+ BN1 stats)
    gemm_kernel<128, 256, false, false, true>
        <<<dim3(gridM, 4), 256, 0, stream>>>(x, w1, nullptr, nullptr, b1, h, sum1, N);
    finalize_bn<<<1, 256, 0, stream>>>(sum1, gamma1, beta1, scale1, shift1, 256, 1.0f / N);
    affine_inplace<<<2048, 256, 0, stream>>>(h, scale1, shift1, NH / 4, H / 4);

    // segment_sum of h over edges
    scatter_kernel<<<4096, 256, 0, stream>>>(h, src, dst, agg, E);

    // GraphConv: g = gelu(h @ w_root^T + agg @ w_nei^T + b_gc)
    gemm_kernel<256, 256, true, true, false>
        <<<dim3(gridM, 4), 256, 0, stream>>>(h, w_root, agg, w_nei, b_gc, g, nullptr, N);

    // fc2 (+ BN2 stats); y_pre overwrites h
    gemm_kernel<256, 128, false, false, true>
        <<<dim3(gridM, 2), 256, 0, stream>>>(g, w2, nullptr, nullptr, b2, ypre, sum2, N);
    finalize_bn<<<1, 128, 0, stream>>>(sum2, gamma2, beta2, scale2, shift2, 128, 1.0f / N);

    // y = BN2(y_pre) + x
    final_residual<<<2048, 256, 0, stream>>>(ypre, x, scale2, shift2, (float*)d_out, ND / 4, D / 4);
}
